// VectorQuantizer_80444737454571
// MI455X (gfx1250) — matrix operand order repaired and checked by reading, not compiled, not run on hardware
//
#include <hip/hip_runtime.h>
#include <hip/hip_bf16.h>

#define NUM_EMB 8192
#define DIM     512
#define NROWS   16384  // 16*1024
#define KSPLIT  4
#define KCHUNK  (NUM_EMB / KSPLIT)   // 2048 codes scanned per block

typedef __attribute__((ext_vector_type(16))) __bf16 v16bf;
typedef __attribute__((ext_vector_type(8)))  float  v8f;
typedef __attribute__((ext_vector_type(4)))  unsigned int v4u;
typedef __attribute__((ext_vector_type(8)))  int v8i;
typedef __attribute__((ext_vector_type(4)))  int v4i;

__device__ __forceinline__ unsigned short f32_to_bf16(float f) {
    unsigned int u = __float_as_uint(f);
    u += 0x7FFFu + ((u >> 16) & 1u);       // round to nearest even
    return (unsigned short)(u >> 16);
}

// ---- kernel 1: emb f32 -> bf16 + row L2 norms ----------------------------
__global__ void __launch_bounds__(256) prep_emb(const float* __restrict__ emb,
                                                unsigned short* __restrict__ embbf,
                                                float* __restrict__ enorm) {
    __shared__ float red[256];
    const int r = blockIdx.x;
    const float* er = emb + (size_t)r * DIM;
    unsigned short* eb = embbf + (size_t)r * DIM;
    float s = 0.f;
    for (int d = threadIdx.x; d < DIM; d += 256) {
        float v = er[d];
        s += v * v;
        eb[d] = f32_to_bf16(v);
    }
    red[threadIdx.x] = s;
    __syncthreads();
    for (int off = 128; off > 0; off >>= 1) {
        if (threadIdx.x < off) red[threadIdx.x] += red[threadIdx.x + off];
        __syncthreads();
    }
    if (threadIdx.x == 0) enorm[r] = red[0];
}

// ---- zero helpers --------------------------------------------------------
__global__ void zero_f2(float2* __restrict__ p, unsigned long long n2) {
    unsigned long long i = (unsigned long long)blockIdx.x * blockDim.x + threadIdx.x;
    unsigned long long stride = (unsigned long long)gridDim.x * blockDim.x;
    const float2 z = {0.f, 0.f};
    for (; i < n2; i += stride) p[i] = z;
}

__global__ void zero_ws(unsigned int* __restrict__ counts, float* __restrict__ lossacc) {
    int i = blockIdx.x * blockDim.x + threadIdx.x;
    if (i < NUM_EMB) counts[i] = 0u;
    if (i == 0) *lossacc = 0.f;
}

// ---- kernel 2: fused bf16-WMMA GEMM + argmax over a codebook slice -------
// block = 256 threads (8 waves); each wave owns 16 rows; block owns 128 rows
// and scans KCHUNK codes. B chunks staged by the Tensor Data Mover into a
// double-buffered LDS tile while the waves compute on the other buffer.
__global__ void __launch_bounds__(256) vq_argmax(const float* __restrict__ x,
                                                 const unsigned short* __restrict__ embbf,
                                                 const float* __restrict__ enorm,
                                                 float* __restrict__ pval,
                                                 int* __restrict__ pidx) {
    __shared__ unsigned short ldsB[2][64 * DIM];   // 2 x 64 KB double buffer

    const int tid  = threadIdx.x;
    const int wave = tid >> 5;
    const int lane = tid & 31;
    const int ln   = lane & 15;   // N (column) index within 16-wide tile
    const int hi   = lane >> 4;   // half-wave selector (K-split per ISA layout)

    const int rb   = blockIdx.x >> 2;        // row block
    const int ks   = blockIdx.x & 3;         // codebook slice
    const int base_m = rb * 128 + wave * 16;
    const int kbeg = ks * KCHUNK;

    // ---- load A (16 rows x 512) into registers in contiguous-K order (k = 16*hi + s) ----
    v16bf a[16];
    {
        const float* xrow = x + (size_t)(base_m + ln) * DIM;
        #pragma unroll
        for (int c = 0; c < 16; ++c) {
            union { v16bf v; unsigned short s[16]; } u;
            const int d0 = c * 32 + hi * 16;
            const int d1 = c * 32 + hi * 16 + 8;
            #pragma unroll
            for (int j = 0; j < 8; ++j) {
                u.s[j]     = f32_to_bf16(xrow[d0 + j]);
                u.s[8 + j] = f32_to_bf16(xrow[d1 + j]);
            }
            a[c] = u.v;
        }
    }

    // LDS group-segment byte offset of the buffers (aperture keeps it in [31:0])
    const unsigned int lds_base = (unsigned int)(uintptr_t)(void*)&ldsB[0][0];

    // TDM descriptor issue: copy 64 codebook rows (64x512 bf16 = 64KB tile)
    auto tdm_issue = [&](int bufsel, int k0) {
        unsigned long long ga =
            (unsigned long long)(uintptr_t)(embbf + (size_t)k0 * DIM);
        v4u g0 = { 1u,                                            // count=1 (valid D#)
                   lds_base + (unsigned int)bufsel * (64u * DIM * 2u),
                   (unsigned int)ga,
                   (unsigned int)((ga >> 32) & 0x01FFFFFFu) | (2u << 30) }; // type=2
        v8i g1 = { (int)(1u << 16),        // data_size = 2 bytes
                   (int)(512u << 16),      // tensor_dim0 = 512 (bits 63:48)
                   (int)(64u  << 16),      // tensor_dim1 = 64  (bits 95:80)
                   (int)(512u << 16),      // tile_dim0  = 512  (bits 127:112)
                   64,                     // tile_dim1  = 64   (bits 143:128)
                   512,                    // tensor_dim0_stride = 512
                   0, 0 };
        v4i z4 = { 0, 0, 0, 0 };
        v8i z8 = { 0, 0, 0, 0, 0, 0, 0, 0 };
        __builtin_amdgcn_tensor_load_to_lds(g0, g1, z4, z4, z8, 0);
    };

    if (wave == 0) tdm_issue(0, kbeg);     // prologue: fetch first chunk

    float bestv[8];
    int   besti[8];
    #pragma unroll
    for (int r = 0; r < 8; ++r) { bestv[r] = -3.4e38f; besti[r] = 0; }

    const int nch = KCHUNK / 64;           // 32 chunks of 64 codes
    for (int i = 0; i < nch; ++i) {
        const int k0 = kbeg + i * 64;
        if (wave == 0) {
            if (i + 1 < nch) {             // prefetch next chunk, then wait for current
                tdm_issue((i + 1) & 1, k0 + 64);
                __builtin_amdgcn_s_wait_tensorcnt(1);
            } else {
                __builtin_amdgcn_s_wait_tensorcnt(0);
            }
        }
        __syncthreads();                   // publish buf[i&1] to all waves

        const unsigned short* B = ldsB[i & 1];
        #pragma unroll
        for (int t = 0; t < 4; ++t) {
            // two independent accumulation chains for XDL ILP
            v8f acc0 = {};
            v8f acc1 = {};
            #pragma unroll
            for (int c = 0; c < 16; c += 2) {
                // B 32x16 bf16 layout: lane ln holds column (t*16+ln),
                // K(d) = c*32 + hi*16 .. +15 -> contiguous 32B in LDS
                const v16bf b0 = *(const v16bf*)&B[(size_t)(t * 16 + ln) * DIM
                                                   + c * 32 + hi * 16];
                const v16bf b1 = *(const v16bf*)&B[(size_t)(t * 16 + ln) * DIM
                                                   + (c + 1) * 32 + hi * 16];
                acc0 = __builtin_amdgcn_wmma_f32_16x16x32_bf16(
                         false, a[c],     false, b0, (short)0, acc0, false, false);
                acc1 = __builtin_amdgcn_wmma_f32_16x16x32_bf16(
                         false, a[c + 1], false, b1, (short)0, acc1, false, false);
            }
            // score = x.e - 0.5*||e||^2  (argmax == argmin of squared distance)
            const int   kcol     = k0 + t * 16 + ln;
            const float halfnorm = 0.5f * enorm[kcol];   // L2-resident table
            #pragma unroll
            for (int r = 0; r < 8; ++r) {
                float v = (acc0[r] + acc1[r]) - halfnorm;
                if (v > bestv[r]) { bestv[r] = v; besti[r] = kcol; }  // ties keep lower k
            }
        }
        __syncthreads();                   // all waves done with buf[i&1]
    }

    // reduce over the 16 lanes sharing a row group; prefer lower index on ties
    #pragma unroll
    for (int r = 0; r < 8; ++r) {
        #pragma unroll
        for (int m = 8; m >= 1; m >>= 1) {
            float ov = __shfl_xor(bestv[r], m, 32);
            int   oi = __shfl_xor(besti[r], m, 32);
            if (ov > bestv[r] || (ov == bestv[r] && oi < besti[r])) {
                bestv[r] = ov; besti[r] = oi;
            }
        }
        // lanes 0..15 hold rows 0..7, lanes 16..31 hold rows 8..15 (C layout)
        if (ln == 0) {
            const size_t o = (size_t)ks * NROWS + base_m + hi * 8 + r;
            pval[o] = bestv[r];
            pidx[o] = besti[r];
        }
    }
}

// ---- kernel 3: merge the KSPLIT partial argmaxes -------------------------
__global__ void __launch_bounds__(256) vq_combine(const float* __restrict__ pval,
                                                  const int* __restrict__ pidx,
                                                  int* __restrict__ out_idx) {
    int n = blockIdx.x * 256 + threadIdx.x;
    if (n >= NROWS) return;
    float bv = pval[n];
    int   bi = pidx[n];
    #pragma unroll
    for (int p = 1; p < KSPLIT; ++p) {
        float v = pval[(size_t)p * NROWS + n];
        int   i = pidx[(size_t)p * NROWS + n];
        if (v > bv || (v == bv && i < bi)) { bv = v; bi = i; }
    }
    out_idx[n] = bi;
}

// ---- kernel 4: gather + loss accum + one-hot scatter + histogram ---------
__global__ void __launch_bounds__(256) vq_assign(const float* __restrict__ x,
                                                 const float* __restrict__ emb,
                                                 const int* __restrict__ idx,
                                                 float* __restrict__ quant,
                                                 float* __restrict__ enc,
                                                 float* __restrict__ idxf,
                                                 unsigned int* __restrict__ counts,
                                                 float* __restrict__ lossacc) {
    __shared__ float red[256];
    const int n = blockIdx.x;
    const int k = idx[n];
    const float* er = emb + (size_t)k * DIM;
    const float* xr = x   + (size_t)n * DIM;
    float* qr = quant + (size_t)n * DIM;
    float s = 0.f;
    for (int d = threadIdx.x; d < DIM; d += 256) {
        float e  = er[d];
        float dv = e - xr[d];
        qr[d] = e;                 // straight-through: x + (q - x) == q
        s += dv * dv;
    }
    red[threadIdx.x] = s;
    __syncthreads();
    for (int off = 128; off > 0; off >>= 1) {
        if (threadIdx.x < off) red[threadIdx.x] += red[threadIdx.x + off];
        __syncthreads();
    }
    if (threadIdx.x == 0) {
        atomicAdd(lossacc, red[0]);
        atomicAdd(&counts[k], 1u);
        enc[(size_t)n * NUM_EMB + k] = 1.0f;
        idxf[n] = (float)k;
    }
}

// ---- kernel 5: finalize loss + perplexity --------------------------------
__global__ void __launch_bounds__(256) vq_finalize(const unsigned int* __restrict__ counts,
                                                   const float* __restrict__ lossacc,
                                                   float* __restrict__ loss_out,
                                                   float* __restrict__ perp_out) {
    __shared__ float red[256];
    float h = 0.f;
    for (int k = threadIdx.x; k < NUM_EMB; k += 256) {
        float p = (float)counts[k] * (1.0f / (float)NROWS);
        h += p * logf(p + 1e-10f);
    }
    red[threadIdx.x] = h;
    __syncthreads();
    for (int off = 128; off > 0; off >>= 1) {
        if (threadIdx.x < off) red[threadIdx.x] += red[threadIdx.x + off];
        __syncthreads();
    }
    if (threadIdx.x == 0) {
        // loss = q_latent + 0.25*e_latent = 1.25 * mean((q-x)^2)
        *loss_out = lossacc[0] * 1.25f / ((float)NROWS * (float)DIM);
        *perp_out = expf(-red[0]);
    }
}

extern "C" void kernel_launch(void* const* d_in, const int* in_sizes, int n_in,
                              void* d_out, int out_size, void* d_ws, size_t ws_size,
                              hipStream_t stream) {
    const float* x   = (const float*)d_in[0];   // [16,1024,512] f32
    const float* emb = (const float*)d_in[1];   // [8192,512]    f32
    float* out = (float*)d_out;

    // output layout: loss | quantized_st | perplexity | encodings | indices
    const size_t QUANT_OFF = 1;
    const size_t PERP_OFF  = 1 + (size_t)NROWS * DIM;            // 8388609
    const size_t ENC_OFF   = PERP_OFF + 1;                       // 8388610
    const size_t IDX_OFF   = ENC_OFF + (size_t)NROWS * NUM_EMB;  // 142606338

    // workspace layout (~9.1 MB)
    char* w = (char*)d_ws;
    unsigned short* embbf  = (unsigned short*)w;   w += (size_t)NUM_EMB * DIM * 2;   // 8 MB
    float*          enorm  = (float*)w;            w += (size_t)NUM_EMB * sizeof(float);
    float*          pval   = (float*)w;            w += (size_t)KSPLIT * NROWS * sizeof(float);
    int*            pidx   = (int*)w;              w += (size_t)KSPLIT * NROWS * sizeof(int);
    int*            idx    = (int*)w;              w += (size_t)NROWS * sizeof(int);
    unsigned int*   counts = (unsigned int*)w;     w += (size_t)NUM_EMB * sizeof(unsigned int);
    float*          lossac = (float*)w;

    prep_emb<<<NUM_EMB, 256, 0, stream>>>(emb, embbf, enorm);
    // encodings offset is 8B-aligned -> float2 zero fill of 537 MB region
    zero_f2<<<8192, 256, 0, stream>>>((float2*)(out + ENC_OFF),
                                      (unsigned long long)NROWS * NUM_EMB / 2ull);
    zero_ws<<<(NUM_EMB + 255) / 256, 256, 0, stream>>>(counts, lossac);
    vq_argmax<<<(NROWS / 128) * KSPLIT, 256, 0, stream>>>(x, embbf, enorm, pval, pidx);
    vq_combine<<<NROWS / 256, 256, 0, stream>>>(pval, pidx, idx);
    vq_assign<<<NROWS, 256, 0, stream>>>(x, emb, idx, out + QUANT_OFF, out + ENC_OFF,
                                         out + IDX_OFF, counts, lossac);
    vq_finalize<<<1, 256, 0, stream>>>(counts, lossac, out + 0, out + PERP_OFF);
}
